// Super_MultiMamba_41154376630475
// MI455X (gfx1250) — compile-verified, matrix operand order checked by reading, not run on hardware
//
#include <hip/hip_runtime.h>
#include <hip/hip_bf16.h>

// ---------------------------------------------------------------------------
// Multi-directional Mamba block (Vim-style 4-direction scan) for MI455X/gfx1250
//   B=4, E=768, D=1536, N=16, R=48, K=4, L=256 (16x16), NDIR=4
// GEMMs use v_wmma_f32_16x16x32_f16 (f16 in, f32 acc), 16x64 output per wave
// (4 accumulators -> 4 independent WMMAs per A fragment). Scan is scalar f32.
// ---------------------------------------------------------------------------

#define Bc    4
#define Ec    768
#define Dc    1536
#define Nc    16
#define Rc    48
#define Kc    4
#define Lc    256
#define NDIRc 4
#define KDIMc 80   // R + 2N

typedef __attribute__((ext_vector_type(8)))  _Float16 half8;
typedef __attribute__((ext_vector_type(16))) _Float16 v16h;
typedef __attribute__((ext_vector_type(8)))  float    v8f;

__device__ __forceinline__ int permL(int dir, int t) {
    // PERMS from reference: identity, reverse, 16x16-transpose, reverse-transpose
    switch (dir) {
        case 0:  return t;
        case 1:  return (Lc - 1) - t;
        case 2:  return ((t & 15) << 4) | (t >> 4);
        default: { int u = (Lc - 1) - t; return ((u & 15) << 4) | (u >> 4); }
    }
}

__device__ __forceinline__ float silu_f(float x) {
    return x / (1.0f + __expf(-x));
}

// ---------------------------------------------------------------------------
// Generic WMMA GEMM:  C[z][m][n] = sum_k A[z][m][k] * B[z][n*ldb + k]
//   A: f16 row-major, addr = (z/aZdiv)*aZ + m*lda + k
//   B: f16 k-contiguous, addr = z*bZ + n*ldb + k
//   C: f32,             addr = z*cZ + m*cM + n*cN
// blockDim = (32, 8): each wave computes a 16(m) x 64(n) strip via 4
// accumulators, reusing one A fragment across 4 WMMAs. K multiple of 32.
// grid.x = Ncol/64.
// ---------------------------------------------------------------------------
__global__ void wmma_gemm_kernel(const _Float16* __restrict__ Ab,
                                 const _Float16* __restrict__ Bb,
                                 float* __restrict__ Cb,
                                 int M, int K,
                                 int lda, int ldb, int cM, int cN,
                                 long aZ, int aZdiv, long bZ, long cZ)
{
    const int lane    = threadIdx.x;        // 0..31
    const int mt      = blockIdx.y * blockDim.y + threadIdx.y;
    if (mt * 16 >= M) return;
    const int nt      = blockIdx.x;         // 64-wide n strip
    const int z       = blockIdx.z;
    const int half_id = lane >> 4;          // 0: lanes 0-15, 1: lanes 16-31
    const int mr      = lane & 15;          // A-row / B-col / D-col for this lane

    const _Float16* __restrict__ arow =
        Ab + (long)(z / aZdiv) * aZ + ((long)mt * 16 + mr) * (long)lda;
    const _Float16* __restrict__ bbase = Bb + (long)z * bZ;

    const _Float16* __restrict__ brow0 = bbase + ((long)nt * 64 +  0 + mr) * (long)ldb;
    const _Float16* __restrict__ brow1 = bbase + ((long)nt * 64 + 16 + mr) * (long)ldb;
    const _Float16* __restrict__ brow2 = bbase + ((long)nt * 64 + 32 + mr) * (long)ldb;
    const _Float16* __restrict__ brow3 = bbase + ((long)nt * 64 + 48 + mr) * (long)ldb;

    v8f acc0 = {}, acc1 = {}, acc2 = {}, acc3 = {};
    for (int k0 = 0; k0 < K; k0 += 32) {
        // 16-bit A 16x32 layout: lanes0-15 K=[0..7|16..23], lanes16-31 K=[8..15|24..31]
        half8 alo = *(const half8*)(arow + k0 +      half_id * 8);
        half8 ahi = *(const half8*)(arow + k0 + 16 + half_id * 8);
        v16h af = __builtin_shufflevector(alo, ahi, 0,1,2,3,4,5,6,7,8,9,10,11,12,13,14,15);

        half8 b0l = *(const half8*)(brow0 + k0 +      half_id * 8);
        half8 b0h = *(const half8*)(brow0 + k0 + 16 + half_id * 8);
        half8 b1l = *(const half8*)(brow1 + k0 +      half_id * 8);
        half8 b1h = *(const half8*)(brow1 + k0 + 16 + half_id * 8);
        half8 b2l = *(const half8*)(brow2 + k0 +      half_id * 8);
        half8 b2h = *(const half8*)(brow2 + k0 + 16 + half_id * 8);
        half8 b3l = *(const half8*)(brow3 + k0 +      half_id * 8);
        half8 b3h = *(const half8*)(brow3 + k0 + 16 + half_id * 8);

        v16h bf0 = __builtin_shufflevector(b0l, b0h, 0,1,2,3,4,5,6,7,8,9,10,11,12,13,14,15);
        v16h bf1 = __builtin_shufflevector(b1l, b1h, 0,1,2,3,4,5,6,7,8,9,10,11,12,13,14,15);
        v16h bf2 = __builtin_shufflevector(b2l, b2h, 0,1,2,3,4,5,6,7,8,9,10,11,12,13,14,15);
        v16h bf3 = __builtin_shufflevector(b3l, b3h, 0,1,2,3,4,5,6,7,8,9,10,11,12,13,14,15);

        acc0 = __builtin_amdgcn_wmma_f32_16x16x32_f16(false, af, false, bf0, (short)0, acc0, false, false);
        acc1 = __builtin_amdgcn_wmma_f32_16x16x32_f16(false, af, false, bf1, (short)0, acc1, false, false);
        acc2 = __builtin_amdgcn_wmma_f32_16x16x32_f16(false, af, false, bf2, (short)0, acc2, false, false);
        acc3 = __builtin_amdgcn_wmma_f32_16x16x32_f16(false, af, false, bf3, (short)0, acc3, false, false);
    }

    // D layout: VGPR j -> row (8*half_id + j), col mr (within each 16-col tile)
    float* __restrict__ crow = Cb + (long)z * cZ + ((long)mt * 16 + 8 * half_id) * (long)cM;
    #pragma unroll
    for (int j = 0; j < 8; ++j) {
        float* cj = crow + (long)j * cM;
        cj[((long)nt * 64 +  0 + mr) * (long)cN] = acc0[j];
        cj[((long)nt * 64 + 16 + mr) * (long)cN] = acc1[j];
        cj[((long)nt * 64 + 32 + mr) * (long)cN] = acc2[j];
        cj[((long)nt * 64 + 48 + mr) * (long)cN] = acc3[j];
    }
}

// ---------------------------------------------------------------------------
// Cast helpers
// ---------------------------------------------------------------------------
__global__ void cast_f32_f16_kernel(const float* __restrict__ src,
                                    _Float16* __restrict__ dst, long n)
{
    long i = (long)blockIdx.x * blockDim.x + threadIdx.x;
    if (i < n) dst[i] = (_Float16)src[i];
}

// dtproj_w [NDIR][D][R=48] -> padded f16 [NDIR][D][64]
__global__ void cast_dtw_pad_kernel(const float* __restrict__ dtw,
                                    _Float16* __restrict__ dst)
{
    long i = (long)blockIdx.x * blockDim.x + threadIdx.x;   // NDIR*D*64
    if (i >= (long)NDIRc * Dc * 64) return;
    int k  = (int)(i & 63);
    long r = i >> 6;                                        // dir*D + d
    dst[i] = (k < Rc) ? (_Float16)dtw[r * Rc + k] : (_Float16)0.0f;
}

// x_dbl f32 [zb][80][L] rows 0..47 -> dt16 f16 [zb][L][64] (zero-padded K)
__global__ void cast_dt_pad_kernel(const float* __restrict__ x_dbl,
                                   _Float16* __restrict__ dt16)
{
    long i = (long)blockIdx.x * blockDim.x + threadIdx.x;   // 16*L*64
    if (i >= (long)NDIRc * Bc * Lc * 64) return;
    int k  = (int)(i & 63);
    int t  = (int)((i >> 6) & (Lc - 1));
    int zb = (int)(i >> 14);                                // dir*B + b
    float v = (k < Rc) ? x_dbl[((long)zb * KDIMc + k) * Lc + t] : 0.0f;
    dt16[i] = (_Float16)v;
}

// ---------------------------------------------------------------------------
// Depthwise causal conv (K=4) + bias + SiLU, with direction permutation folded
// in. Writes t-major f16 for the x_dbl WMMA GEMM: xconv16[zb][t][d].
// One thread per (dir, b, d).
// ---------------------------------------------------------------------------
__global__ void conv_silu_kernel(const float* __restrict__ xz,
                                 const float* __restrict__ conv_w,
                                 const float* __restrict__ conv_b,
                                 _Float16* __restrict__ xconv16)
{
    int g = blockIdx.x * blockDim.x + threadIdx.x;          // (dir*B+b)*D + d
    if (g >= NDIRc * Bc * Dc) return;
    int d   = g % Dc;
    int zb  = g / Dc;
    int b   = zb % Bc;
    int dir = zb / Bc;

    const float* __restrict__ xrow = xz + ((long)b * 2 * Dc + d) * Lc;
    const float w0 = conv_w[((long)dir * Dc + d) * Kc + 0];
    const float w1 = conv_w[((long)dir * Dc + d) * Kc + 1];
    const float w2 = conv_w[((long)dir * Dc + d) * Kc + 2];
    const float w3 = conv_w[((long)dir * Dc + d) * Kc + 3];
    const float cb = conv_b[(long)dir * Dc + d];

    float x0 = 0.f, x1 = 0.f, x2 = 0.f;
    _Float16* __restrict__ orow = xconv16 + (long)zb * Lc * Dc + d;
    for (int t = 0; t < Lc; ++t) {
        float x3 = xrow[permL(dir, t)];
        float a  = w0 * x0 + w1 * x1 + w2 * x2 + w3 * x3 + cb;
        orow[(long)t * Dc] = (_Float16)silu_f(a);
        x0 = x1; x1 = x2; x2 = x3;
    }
}

// ---------------------------------------------------------------------------
// Selective scan. One thread per (dir, b, d, n); 16 lanes (n) cooperate via
// shfl_xor reduction (wave32-safe). Writes y at original (un-permuted) l.
// ---------------------------------------------------------------------------
__global__ void scan_kernel(const float* __restrict__ delta_pre,
                            const _Float16* __restrict__ xconv16,
                            const float* __restrict__ x_dbl,
                            const float* __restrict__ xz,
                            const float* __restrict__ dtproj_b,
                            const float* __restrict__ A_log,
                            const float* __restrict__ D_param,
                            float* __restrict__ yout)
{
    int g = blockIdx.x * blockDim.x + threadIdx.x;          // ((dir*B+b)*D+d)*16+n
    int n    = g & 15;
    int rest = g >> 4;                                      // (dir*B+b)*D + d
    int d    = rest % Dc;
    int zb   = rest / Dc;
    int b    = zb % Bc;
    int dir  = zb / Bc;

    const float dtb = dtproj_b[(long)dir * Dc + d];
    const float An  = -__expf(A_log[((long)dir * Dc + d) * Nc + n]);
    const float Dp  = D_param[(long)dir * Dc + d];

    const float*    __restrict__ dprow = delta_pre + (long)rest * Lc;
    const _Float16* __restrict__ xcol  = xconv16 + (long)zb * Lc * Dc + d;
    const float*    __restrict__ Brow  = x_dbl + ((long)zb * KDIMc + Rc + n) * Lc;
    const float*    __restrict__ Crow  = x_dbl + ((long)zb * KDIMc + Rc + Nc + n) * Lc;
    const float*    __restrict__ zrow  = xz + ((long)b * 2 * Dc + Dc + d) * Lc;
    float*          __restrict__ yrow  = yout + (long)rest * Lc;

    float h = 0.0f;
    for (int t = 0; t < Lc; ++t) {
        float pre   = dprow[t] + dtb;
        float delta = (pre > 20.0f) ? pre : log1pf(__expf(pre));   // softplus
        float xv    = (float)xcol[(long)t * Dc];
        float da    = __expf(delta * An);
        h = da * h + delta * Brow[t] * xv;
        float part = h * Crow[t];
        part += __shfl_xor(part, 1, 32);
        part += __shfl_xor(part, 2, 32);
        part += __shfl_xor(part, 4, 32);
        part += __shfl_xor(part, 8, 32);
        if (n == 0) {
            int   l  = permL(dir, t);       // inverse perm: out[PERM[t]] = y[t]
            float y  = part + Dp * xv;
            float zv = zrow[l];
            yrow[l]  = y * silu_f(zv);
        }
    }
}

// Sum 4 directions, cast to f16 in [b][l][d] layout for the out_proj GEMM.
__global__ void ysum_kernel(const float* __restrict__ y,
                            _Float16* __restrict__ ysum16)
{
    long i = (long)blockIdx.x * blockDim.x + threadIdx.x;   // (b*L+l)*D + d
    if (i >= (long)Bc * Lc * Dc) return;
    int  d  = (int)(i % Dc);
    long bl = i / Dc;
    int  b  = (int)(bl / Lc);
    int  l  = (int)(bl % Lc);
    float s = 0.f;
    #pragma unroll
    for (int dir = 0; dir < NDIRc; ++dir)
        s += y[(((long)(dir * Bc + b)) * Dc + d) * Lc + l];
    ysum16[i] = (_Float16)s;
}

// ---------------------------------------------------------------------------
// Host-side launcher
// ---------------------------------------------------------------------------
static inline size_t align256(size_t x) { return (x + 255) & ~(size_t)255; }

extern "C" void kernel_launch(void* const* d_in, const int* in_sizes, int n_in,
                              void* d_out, int out_size, void* d_ws, size_t ws_size,
                              hipStream_t stream)
{
    const float* hidden   = (const float*)d_in[0];  // [B][L][E]
    const float* in_w     = (const float*)d_in[1];  // [2D][E]
    const float* out_w    = (const float*)d_in[2];  // [E][D]
    const float* conv_w   = (const float*)d_in[3];  // [NDIR][D][K]
    const float* conv_b   = (const float*)d_in[4];  // [NDIR][D]
    const float* xproj_w  = (const float*)d_in[5];  // [NDIR][80][D]
    const float* dtproj_w = (const float*)d_in[6];  // [NDIR][D][R]
    const float* dtproj_b = (const float*)d_in[7];  // [NDIR][D]
    const float* A_log    = (const float*)d_in[8];  // [NDIR][D][N]
    const float* D_param  = (const float*)d_in[9];  // [NDIR][D]
    float* out            = (float*)d_out;          // [B][L][E]

    // ---- workspace carve-out (bytes) ----
    char* ws = (char*)d_ws;
    size_t off = 0;
    auto take = [&](size_t bytes) { char* p = ws + off; off += align256(bytes); return p; };

    _Float16* hs16    = (_Float16*)take((size_t)Bc * Lc * Ec * 2);            // [B][L][E]
    _Float16* w_in16  = (_Float16*)take((size_t)2 * Dc * Ec * 2);             // [2D][E]
    _Float16* xw16    = (_Float16*)take((size_t)NDIRc * KDIMc * Dc * 2);      // [dir][80][D]
    _Float16* dtw16   = (_Float16*)take((size_t)NDIRc * Dc * 64 * 2);         // [dir][D][64]
    _Float16* ow16    = (_Float16*)take((size_t)Ec * Dc * 2);                 // [E][D]
    float*    xz      = (float*)take((size_t)Bc * 2 * Dc * Lc * 4);           // [B][2D][L]
    _Float16* xconv16 = (_Float16*)take((size_t)NDIRc * Bc * Lc * Dc * 2);    // [zb][t][d]
    float*    x_dbl   = (float*)take((size_t)NDIRc * Bc * KDIMc * Lc * 4);    // [zb][80][L]
    _Float16* dt16    = (_Float16*)take((size_t)NDIRc * Bc * Lc * 64 * 2);    // [zb][t][64]
    float*    dpre    = (float*)take((size_t)NDIRc * Bc * Dc * Lc * 4);       // [zb][d][t]
    float*    ybuf    = (float*)take((size_t)NDIRc * Bc * Dc * Lc * 4);       // [zb][d][l]
    _Float16* ysum16  = (_Float16*)take((size_t)Bc * Lc * Dc * 2);            // [b][l][d]
    (void)ws_size; (void)in_sizes; (void)n_in; (void)out_size;

    const dim3 gblk(32, 8, 1);   // 8 waves/block, one 16x64 strip per wave

    // ---- stage f16 operands ----
    {
        long n;
        n = (long)Bc * Lc * Ec;
        cast_f32_f16_kernel<<<(n + 255) / 256, 256, 0, stream>>>(hidden, hs16, n);
        n = (long)2 * Dc * Ec;
        cast_f32_f16_kernel<<<(n + 255) / 256, 256, 0, stream>>>(in_w, w_in16, n);
        n = (long)NDIRc * KDIMc * Dc;
        cast_f32_f16_kernel<<<(n + 255) / 256, 256, 0, stream>>>(xproj_w, xw16, n);
        n = (long)Ec * Dc;
        cast_f32_f16_kernel<<<(n + 255) / 256, 256, 0, stream>>>(out_w, ow16, n);
        n = (long)NDIRc * Dc * 64;
        cast_dtw_pad_kernel<<<(n + 255) / 256, 256, 0, stream>>>(dtproj_w, dtw16);
    }

    // ---- GEMM 1: in_proj  xz[b][f][l] = sum_e W[f][e]*hs[b][l][e] ----
    // M=2D=3072, Ncol=L=256 per b-slice, K=E=768
    wmma_gemm_kernel<<<dim3(Lc / 64, (2 * Dc) / 128, Bc), gblk, 0, stream>>>(
        w_in16, hs16, xz,
        /*M=*/2 * Dc, /*K=*/Ec,
        /*lda=*/Ec, /*ldb=*/Ec, /*cM=*/Lc, /*cN=*/1,
        /*aZ=*/0, /*aZdiv=*/1, /*bZ=*/(long)Lc * Ec, /*cZ=*/(long)2 * Dc * Lc);

    // ---- conv + SiLU (all 4 dirs), writes xconv16[zb][t][d] ----
    {
        int n = NDIRc * Bc * Dc;
        conv_silu_kernel<<<(n + 255) / 256, 256, 0, stream>>>(xz, conv_w, conv_b, xconv16);
    }

    // ---- GEMM 2: x_dbl[zb][k][t] = sum_d xw[dir][k][d]*xconv[zb][t][d] ----
    // M=80, Ncol=L per zb-slice (16 slices), K=D=1536
    wmma_gemm_kernel<<<dim3(Lc / 64, 1, NDIRc * Bc), gblk, 0, stream>>>(
        xw16, xconv16, x_dbl,
        /*M=*/KDIMc, /*K=*/Dc,
        /*lda=*/Dc, /*ldb=*/Dc, /*cM=*/Lc, /*cN=*/1,
        /*aZ=*/(long)KDIMc * Dc, /*aZdiv=*/Bc,
        /*bZ=*/(long)Lc * Dc, /*cZ=*/(long)KDIMc * Lc);

    // ---- stage dt rows (0..47) of x_dbl as padded f16 [zb][t][64] ----
    {
        long n = (long)NDIRc * Bc * Lc * 64;
        cast_dt_pad_kernel<<<(n + 255) / 256, 256, 0, stream>>>(x_dbl, dt16);
    }

    // ---- GEMM 3: delta_pre[zb][d][t] = sum_r dtw[dir][d][r]*dt[zb][t][r] ----
    // M=D=1536, Ncol=L per zb-slice, K=64 (R padded)
    wmma_gemm_kernel<<<dim3(Lc / 64, Dc / 128, NDIRc * Bc), gblk, 0, stream>>>(
        dtw16, dt16, dpre,
        /*M=*/Dc, /*K=*/64,
        /*lda=*/64, /*ldb=*/64, /*cM=*/Lc, /*cN=*/1,
        /*aZ=*/(long)Dc * 64, /*aZdiv=*/Bc,
        /*bZ=*/(long)Lc * 64, /*cZ=*/(long)Dc * Lc);

    // ---- selective scan + gating (all dirs) ----
    {
        int n = NDIRc * Bc * Dc * Nc;   // 393216
        scan_kernel<<<n / 256, 256, 0, stream>>>(
            dpre, xconv16, x_dbl, xz, dtproj_b, A_log, D_param, ybuf);
    }

    // ---- sum directions -> f16 [b][l][d] ----
    {
        long n = (long)Bc * Lc * Dc;
        ysum_kernel<<<(n + 255) / 256, 256, 0, stream>>>(ybuf, ysum16);
    }

    // ---- GEMM 4: out[b][l][e] = sum_d Wout[e][d]*ysum[b][l][d] ----
    // M=E=768, Ncol=L per b-slice, K=D=1536
    wmma_gemm_kernel<<<dim3(Lc / 64, Ec / 128, Bc), gblk, 0, stream>>>(
        ow16, ysum16, out,
        /*M=*/Ec, /*K=*/Dc,
        /*lda=*/Dc, /*ldb=*/Dc, /*cM=*/1, /*cN=*/Ec,
        /*aZ=*/0, /*aZdiv=*/1, /*bZ=*/(long)Lc * Dc, /*cZ=*/(long)Lc * Ec);
}
